// Conv2dSubsamplingFUSED_20_36661840839481
// MI455X (gfx1250) — matrix operand order repaired and checked by reading, not compiled, not run on hardware
//
#include <hip/hip_runtime.h>
#include <math.h>

typedef _Float16 half_t;
typedef __attribute__((ext_vector_type(16))) _Float16 v16h;
typedef __attribute__((ext_vector_type(8)))  _Float16 v8h;
typedef __attribute__((ext_vector_type(8)))  float    v8f;

#define B_     8
#define T_     2048
#define IDIM_  80
#define ODIM_  256
#define T1_    1023
#define F1_    39
#define T2_    511
#define F2_    19
#define NREPR_ 4
#define NEXP_  8
#define KC2_   2304      // conv2 GEMM K  (= 9*256)
#define KT2_   72        // KC2/32
#define KLIN_  4864      // linear GEMM K (= 19*256)
#define KTLIN_ 152
#define KEXP_  1024      // expert GEMM K (= 4*256)
#define KTEXP_ 32
#define NPOS2_ (B_*T2_*F2_)   // 77672 conv2 output positions
#define NTOK_  (B_*T2_)       // 4088 tokens

// ---------------------------------------------------------------------------
// WMMA helpers (wave32, 16x16x32 f16 -> f32)
// ---------------------------------------------------------------------------
__device__ __forceinline__ v8f wmma_f16(v16h a, v16h b, v8f c) {
  // (neg_a, A, neg_b, B, c_mod, C, reuse_a, reuse_b)
  return __builtin_amdgcn_wmma_f32_16x16x32_f16(false, a, false, b, (short)0, c,
                                                false, false);
}

// A-fragment: 16-bit A 16x32 layout. Lane L (0..15) row M=L holds K {0..7,16..23};
// lane L+16 same row holds K {8..15,24..31}. aoff = (lane>=16)?8:0.
__device__ __forceinline__ v16h load_a_frag(const half_t* base, int aoff) {
  v8h lo = *(const v8h*)(base + aoff);
  v8h hi = *(const v8h*)(base + aoff + 16);
  return __builtin_shufflevector(lo, hi, 0, 1, 2, 3, 4, 5, 6, 7,
                                 8, 9, 10, 11, 12, 13, 14, 15);
}

// ---------------------------------------------------------------------------
// conv1: 1 input channel, 3x3 stride 2, ReLU, write f16 channels-last
// h1 layout: [b][t1=1023][f1=39][c=256]
// ---------------------------------------------------------------------------
__global__ void conv1_kernel(const float* __restrict__ x,
                             const float* __restrict__ w,
                             const float* __restrict__ bias,
                             half_t* __restrict__ h1) {
  int pos = blockIdx.x;                 // b*T1*F1 + tq*F1 + fq
  int o = threadIdx.x;                  // output channel
  int b = pos / (T1_ * F1_);
  int r = pos % (T1_ * F1_);
  int tq = r / F1_, fq = r % F1_;
  const float* xb = x + ((size_t)b * T_ + 2 * tq) * IDIM_ + 2 * fq;
  const float* wo = w + o * 9;
  float acc = bias[o];
#pragma unroll
  for (int kh = 0; kh < 3; ++kh)
#pragma unroll
    for (int kw = 0; kw < 3; ++kw)
      acc += xb[kh * IDIM_ + kw] * wo[kh * 3 + kw];
  acc = acc > 0.f ? acc : 0.f;
  h1[(size_t)pos * ODIM_ + o] = (half_t)acc;
}

// ---------------------------------------------------------------------------
// Pack conv2 weights into WMMA B-fragment order, K = (kh*3+kw)*256 + ci
// frag[((ntile*KT2 + ktile)*32 + lane)*16 + j]
// B 16-bit layout: lane 0..15 = column N=lane, K {k0..k0+7,k0+16..k0+23};
//                  lane 16..31 = column N=lane-16, K {k0+8..k0+15,k0+24..k0+31}
// ---------------------------------------------------------------------------
__global__ void pack_w2_frag(const float* __restrict__ w2,
                             half_t* __restrict__ frag) {
  int idx = blockIdx.x * blockDim.x + threadIdx.x;
  if (idx >= 16 * KT2_ * 512) return;
  int j = idx & 15;
  int lane = (idx >> 4) & 31;
  int ktile = (idx >> 9) % KT2_;
  int ntile = idx / (KT2_ * 512);
  int k = ktile * 32 + ((lane >= 16) ? 8 : 0) + (j & 7) + ((j >= 8) ? 16 : 0);
  int n = ntile * 16 + (lane & 15);
  int seg = k >> 8;                       // kh*3+kw
  int ci = k & 255;
  int kh = seg / 3, kw = seg % 3;
  frag[idx] = (half_t)w2[(((size_t)n * ODIM_ + ci) * 3 + kh) * 3 + kw];
}

// Pack row-major [N][K] fp32 weight matrix into WMMA B-fragment f16 order.
__global__ void pack_b_rowmajor(const float* __restrict__ src,
                                half_t* __restrict__ frag, int K, int KT) {
  int idx = blockIdx.x * blockDim.x + threadIdx.x;
  if (idx >= 16 * KT * 512) return;
  int j = idx & 15;
  int lane = (idx >> 4) & 31;
  int ktile = (idx >> 9) % KT;
  int ntile = idx / (KT * 512);
  int k = ktile * 32 + ((lane >= 16) ? 8 : 0) + (j & 7) + ((j >= 8) ? 16 : 0);
  int n = ntile * 16 + (lane & 15);
  frag[idx] = (half_t)src[(size_t)n * K + k];
}

// ---------------------------------------------------------------------------
// conv2 as implicit GEMM: M = 77672 positions, N = 256, K = 2304
// Register-blocked: one wave computes a 32x64 tile (2 M-sub x 4 N-sub,
// 8 accumulators). 8 waves/block -> block covers 64 M x 256 N.
// Per K-step: 4 A-b128 + 8 B-b128 loads feed 8 WMMAs (1.5 loads/WMMA).
// ---------------------------------------------------------------------------
__global__ void __launch_bounds__(256)
conv2_wmma_kernel(const half_t* __restrict__ h1,
                  const half_t* __restrict__ frag,
                  const float* __restrict__ bias,
                  half_t* __restrict__ h2) {
  int lane = threadIdx.x & 31;
  int wave = threadIdx.x >> 5;
  int msub = wave >> 2;                 // 0..1
  int n4   = wave & 3;                  // 0..3 -> ntiles n4*4 .. n4*4+3
  int mbase = blockIdx.x * 64 + msub * 32;
  int aoff = (lane >= 16) ? 8 : 0;

  size_t base00[2];
#pragma unroll
  for (int s = 0; s < 2; ++s) {
    int row = mbase + s * 16 + (lane & 15);
    if (row >= NPOS2_) row = NPOS2_ - 1;       // clamp tail tile loads
    int b = row / (T2_ * F2_);
    int r = row % (T2_ * F2_);
    int tq = r / F2_, fq = r % F2_;
    base00[s] = ((size_t)(b * T1_ + 2 * tq) * F1_ + 2 * fq) * ODIM_;
  }

  const half_t* Bb0 = frag + (size_t)(n4 * 4 + 0) * (KT2_ * 512) + lane * 16;
  const half_t* Bb1 = frag + (size_t)(n4 * 4 + 1) * (KT2_ * 512) + lane * 16;
  const half_t* Bb2 = frag + (size_t)(n4 * 4 + 2) * (KT2_ * 512) + lane * 16;
  const half_t* Bb3 = frag + (size_t)(n4 * 4 + 3) * (KT2_ * 512) + lane * 16;

  v8f acc[2][4] = {};
  for (int seg = 0; seg < 9; ++seg) {
    int kh = seg / 3, kw = seg % 3;
    size_t soff = (size_t)(kh * F1_ + kw) * ODIM_;
    const half_t* A0 = h1 + base00[0] + soff;
    const half_t* A1 = h1 + base00[1] + soff;
    __builtin_prefetch(Bb0 + (seg * 8 + 8) * 512, 0, 1);
    __builtin_prefetch(Bb2 + (seg * 8 + 8) * 512, 0, 1);
#pragma unroll
    for (int ki = 0; ki < 8; ++ki) {
      int ks = seg * 8 + ki;
      v16h a0 = load_a_frag(A0 + ki * 32, aoff);
      v16h a1 = load_a_frag(A1 + ki * 32, aoff);
      v16h b0 = *(const v16h*)(Bb0 + ks * 512);
      v16h b1 = *(const v16h*)(Bb1 + ks * 512);
      v16h b2 = *(const v16h*)(Bb2 + ks * 512);
      v16h b3 = *(const v16h*)(Bb3 + ks * 512);
      acc[0][0] = wmma_f16(a0, b0, acc[0][0]);
      acc[0][1] = wmma_f16(a0, b1, acc[0][1]);
      acc[0][2] = wmma_f16(a0, b2, acc[0][2]);
      acc[0][3] = wmma_f16(a0, b3, acc[0][3]);
      acc[1][0] = wmma_f16(a1, b0, acc[1][0]);
      acc[1][1] = wmma_f16(a1, b1, acc[1][1]);
      acc[1][2] = wmma_f16(a1, b2, acc[1][2]);
      acc[1][3] = wmma_f16(a1, b3, acc[1][3]);
    }
  }

#pragma unroll
  for (int s = 0; s < 2; ++s) {
    int mb = mbase + s * 16 + ((lane >= 16) ? 8 : 0);
#pragma unroll
    for (int t = 0; t < 4; ++t) {
      int n = (n4 * 4 + t) * 16 + (lane & 15);
      float bv = bias[n];
#pragma unroll
      for (int r8 = 0; r8 < 8; ++r8) {
        int m = mb + r8;
        if (m < NPOS2_) {
          float v = acc[s][t][r8] + bv;
          v = v > 0.f ? v : 0.f;
          h2[(size_t)m * ODIM_ + n] = (half_t)v;   // [b][t][f][c] == lin K-order
        }
      }
    }
  }
}

// ---------------------------------------------------------------------------
// linear: M = 4088 tokens, K = 4864, N = 256; same 32x64 register blocking.
// Scale by sqrt(256)=16, write f16 into concat slot [token][tower*256 + n].
// ---------------------------------------------------------------------------
__global__ void __launch_bounds__(256)
lin_wmma_kernel(const half_t* __restrict__ h2,
                const half_t* __restrict__ frag,
                const float* __restrict__ bias,
                half_t* __restrict__ xcat, int tower) {
  int lane = threadIdx.x & 31;
  int wave = threadIdx.x >> 5;
  int msub = wave >> 2;
  int n4   = wave & 3;
  int mbase = blockIdx.x * 64 + msub * 32;
  int aoff = (lane >= 16) ? 8 : 0;

  const half_t* Arow[2];
#pragma unroll
  for (int s = 0; s < 2; ++s) {
    int row = mbase + s * 16 + (lane & 15);
    if (row >= NTOK_) row = NTOK_ - 1;
    Arow[s] = h2 + (size_t)row * KLIN_;
  }
  const half_t* Bb0 = frag + (size_t)(n4 * 4 + 0) * (KTLIN_ * 512) + lane * 16;
  const half_t* Bb1 = frag + (size_t)(n4 * 4 + 1) * (KTLIN_ * 512) + lane * 16;
  const half_t* Bb2 = frag + (size_t)(n4 * 4 + 2) * (KTLIN_ * 512) + lane * 16;
  const half_t* Bb3 = frag + (size_t)(n4 * 4 + 3) * (KTLIN_ * 512) + lane * 16;

  v8f acc[2][4] = {};
#pragma unroll 2
  for (int ks = 0; ks < KTLIN_; ++ks) {
    __builtin_prefetch(Bb0 + (ks + 4) * 512, 0, 1);
    v16h a0 = load_a_frag(Arow[0] + ks * 32, aoff);
    v16h a1 = load_a_frag(Arow[1] + ks * 32, aoff);
    v16h b0 = *(const v16h*)(Bb0 + ks * 512);
    v16h b1 = *(const v16h*)(Bb1 + ks * 512);
    v16h b2 = *(const v16h*)(Bb2 + ks * 512);
    v16h b3 = *(const v16h*)(Bb3 + ks * 512);
    acc[0][0] = wmma_f16(a0, b0, acc[0][0]);
    acc[0][1] = wmma_f16(a0, b1, acc[0][1]);
    acc[0][2] = wmma_f16(a0, b2, acc[0][2]);
    acc[0][3] = wmma_f16(a0, b3, acc[0][3]);
    acc[1][0] = wmma_f16(a1, b0, acc[1][0]);
    acc[1][1] = wmma_f16(a1, b1, acc[1][1]);
    acc[1][2] = wmma_f16(a1, b2, acc[1][2]);
    acc[1][3] = wmma_f16(a1, b3, acc[1][3]);
  }

#pragma unroll
  for (int s = 0; s < 2; ++s) {
    int mb = mbase + s * 16 + ((lane >= 16) ? 8 : 0);
#pragma unroll
    for (int t = 0; t < 4; ++t) {
      int n = (n4 * 4 + t) * 16 + (lane & 15);
      float bv = bias[n];
#pragma unroll
      for (int r8 = 0; r8 < 8; ++r8) {
        int m = mb + r8;
        if (m < NTOK_)
          xcat[(size_t)m * (NREPR_ * ODIM_) + tower * ODIM_ + n] =
              (half_t)(16.0f * (acc[s][t][r8] + bv));
      }
    }
  }
}

// ---------------------------------------------------------------------------
// gate logits (tiny GEMV) + top-2 softmax weights
// ---------------------------------------------------------------------------
__global__ void gate_kernel(const half_t* __restrict__ x,
                            const float* __restrict__ gw,
                            const float* __restrict__ gb,
                            float* __restrict__ logits) {
  int idx = blockIdx.x * blockDim.x + threadIdx.x;
  if (idx >= NTOK_ * NEXP_) return;
  int t = idx / NEXP_, e = idx % NEXP_;
  const half_t* xr = x + (size_t)t * (NREPR_ * ODIM_);
  const float* wr = gw + (size_t)e * (NREPR_ * ODIM_);
  float acc = gb[e];
  for (int k = 0; k < NREPR_ * ODIM_; ++k) acc += (float)xr[k] * wr[k];
  logits[idx] = acc;
}

__global__ void topk_kernel(const float* __restrict__ logits,
                            float* __restrict__ W) {
  int t = blockIdx.x * blockDim.x + threadIdx.x;
  if (t >= NTOK_) return;
  const float* g = logits + (size_t)t * NEXP_;
  int i0 = 0;
  float v0 = g[0];
#pragma unroll
  for (int e = 1; e < NEXP_; ++e)
    if (g[e] > v0) { v0 = g[e]; i0 = e; }
  int i1 = (i0 == 0) ? 1 : 0;
  float v1 = g[i1];
#pragma unroll
  for (int e = 0; e < NEXP_; ++e)
    if (e != i0 && g[e] > v1) { v1 = g[e]; i1 = e; }
  float e0 = 1.0f;                    // exp(v0 - v0)
  float e1 = __expf(v1 - v0);
  float inv = 1.0f / (e0 + e1);
#pragma unroll
  for (int e = 0; e < NEXP_; ++e) W[(size_t)t * NEXP_ + e] = 0.f;
  W[(size_t)t * NEXP_ + i0] = e0 * inv;
  W[(size_t)t * NEXP_ + i1] = e1 * inv;
}

// ---------------------------------------------------------------------------
// experts: 32x32 tile per wave (2 M-sub x 2 N-sub); run all 8 expert GEMMs
// (K=1024) and fold in the sparse top-2 softmax weights in the epilogue.
// 8 waves/block cover the full N=256; grid.x covers M in 32-row blocks.
// ---------------------------------------------------------------------------
__global__ void __launch_bounds__(256)
expert_wmma_kernel(const half_t* __restrict__ x,
                   const half_t* __restrict__ frag,
                   const float* __restrict__ eb,
                   const float* __restrict__ W,
                   float* __restrict__ out) {
  int lane = threadIdx.x & 31;
  int wave = threadIdx.x >> 5;          // 0..7 -> ntiles wave*2, wave*2+1
  int mbase = blockIdx.x * 32;
  int aoff = (lane >= 16) ? 8 : 0;

  const half_t* Arow[2];
#pragma unroll
  for (int s = 0; s < 2; ++s) {
    int row = mbase + s * 16 + (lane & 15);
    if (row >= NTOK_) row = NTOK_ - 1;
    Arow[s] = x + (size_t)row * KEXP_;
  }

  v8f outacc[2][2] = {};
  for (int e = 0; e < NEXP_; ++e) {
    const half_t* Bb0 =
        frag + ((size_t)e * 16 + wave * 2 + 0) * (KTEXP_ * 512) + lane * 16;
    const half_t* Bb1 =
        frag + ((size_t)e * 16 + wave * 2 + 1) * (KTEXP_ * 512) + lane * 16;
    v8f acc[2][2] = {};
#pragma unroll 4
    for (int ks = 0; ks < KTEXP_; ++ks) {
      v16h a0 = load_a_frag(Arow[0] + ks * 32, aoff);
      v16h a1 = load_a_frag(Arow[1] + ks * 32, aoff);
      v16h b0 = *(const v16h*)(Bb0 + ks * 512);
      v16h b1 = *(const v16h*)(Bb1 + ks * 512);
      acc[0][0] = wmma_f16(a0, b0, acc[0][0]);
      acc[0][1] = wmma_f16(a0, b1, acc[0][1]);
      acc[1][0] = wmma_f16(a1, b0, acc[1][0]);
      acc[1][1] = wmma_f16(a1, b1, acc[1][1]);
    }
#pragma unroll
    for (int s = 0; s < 2; ++s) {
      int mb = mbase + s * 16 + ((lane >= 16) ? 8 : 0);
#pragma unroll
      for (int t = 0; t < 2; ++t) {
        int n = (wave * 2 + t) * 16 + (lane & 15);
        float bv = eb[e * ODIM_ + n];
#pragma unroll
        for (int r8 = 0; r8 < 8; ++r8) {
          int m = mb + r8;
          int mm = (m < NTOK_) ? m : (NTOK_ - 1);
          float wgt = W[(size_t)mm * NEXP_ + e];
          outacc[s][t][r8] += wgt * (acc[s][t][r8] + bv);
        }
      }
    }
  }

#pragma unroll
  for (int s = 0; s < 2; ++s) {
    int mb = mbase + s * 16 + ((lane >= 16) ? 8 : 0);
#pragma unroll
    for (int t = 0; t < 2; ++t) {
      int n = (wave * 2 + t) * 16 + (lane & 15);
#pragma unroll
      for (int r8 = 0; r8 < 8; ++r8) {
        int m = mb + r8;
        if (m < NTOK_) out[(size_t)m * ODIM_ + n] = outacc[s][t][r8];
      }
    }
  }
}

// ---------------------------------------------------------------------------
// positional embedding: pe[p][d], pos = (t-1)-p, interleaved sin/cos
// ---------------------------------------------------------------------------
__global__ void posemb_kernel(float* __restrict__ pe) {
  int idx = blockIdx.x * blockDim.x + threadIdx.x;
  if (idx >= (2 * T2_ - 1) * ODIM_) return;
  int p = idx / ODIM_, d = idx % ODIM_;
  float pos = (float)(T2_ - 1 - p);
  int m = d >> 1;
  float div = __expf((float)(2 * m) * (-9.210340371976184f / (float)ODIM_));
  float ang = pos * div;
  pe[idx] = (d & 1) ? __cosf(ang) : __sinf(ang);
}

__global__ void mask_kernel(const unsigned char* __restrict__ xm,
                            float* __restrict__ mo) {
  int idx = blockIdx.x * blockDim.x + threadIdx.x;
  if (idx >= B_ * T2_) return;
  int b = idx / T2_, j = idx % T2_;
  mo[idx] = xm[(size_t)b * T_ + 4 * j] ? 1.0f : 0.0f;
}

// ---------------------------------------------------------------------------
// launch
// ---------------------------------------------------------------------------
extern "C" void kernel_launch(void* const* d_in, const int* in_sizes, int n_in,
                              void* d_out, int out_size, void* d_ws,
                              size_t ws_size, hipStream_t stream) {
  (void)in_sizes; (void)n_in; (void)out_size; (void)ws_size;
  const float* xs[4] = {(const float*)d_in[0], (const float*)d_in[1],
                        (const float*)d_in[2], (const float*)d_in[3]};
  const unsigned char* xmask = (const unsigned char*)d_in[4];
  const float* c1w = (const float*)d_in[5];
  const float* c1b = (const float*)d_in[6];
  const float* c2w = (const float*)d_in[7];
  const float* c2b = (const float*)d_in[8];
  const float* lw  = (const float*)d_in[9];
  const float* lb  = (const float*)d_in[10];
  const float* gw  = (const float*)d_in[11];
  const float* gb  = (const float*)d_in[12];
  const float* ew  = (const float*)d_in[13];
  const float* ebp = (const float*)d_in[14];

  char* ws = (char*)d_ws;
  half_t* h1     = (half_t*)(ws);                      // 163,418,112 B
  half_t* h2     = (half_t*)(ws + 163418112ull);       //  39,768,064 B
  half_t* xcat   = (half_t*)(ws + 203186176ull);       //   8,372,224 B
  half_t* w2f    = (half_t*)(ws + 211558400ull);       //   1,179,648 B
  half_t* linf   = (half_t*)(ws + 212738048ull);       //   2,490,368 B
  half_t* expfr  = (half_t*)(ws + 215228416ull);       //   4,194,304 B
  float*  logits = (float*)(ws + 219422720ull);        //     130,816 B
  float*  Wsel   = (float*)(ws + 219553536ull);        //     130,816 B

  float* out = (float*)d_out;
  float* pe  = out + (size_t)NTOK_ * ODIM_;            // 1,046,528 floats
  float* mo  = pe + (size_t)(2 * T2_ - 1) * ODIM_;     //   261,376 floats

  for (int i = 0; i < NREPR_; ++i) {
    conv1_kernel<<<B_ * T1_ * F1_, 256, 0, stream>>>(
        xs[i], c1w + (size_t)i * ODIM_ * 9, c1b + i * ODIM_, h1);
    pack_w2_frag<<<(16 * KT2_ * 512 + 255) / 256, 256, 0, stream>>>(
        c2w + (size_t)i * ODIM_ * ODIM_ * 9, w2f);
    dim3 g2((NPOS2_ + 63) / 64, 1);
    conv2_wmma_kernel<<<g2, 256, 0, stream>>>(h1, w2f, c2b + i * ODIM_, h2);
    pack_b_rowmajor<<<(16 * KTLIN_ * 512 + 255) / 256, 256, 0, stream>>>(
        lw + (size_t)i * ODIM_ * KLIN_, linf, KLIN_, KTLIN_);
    dim3 gl((NTOK_ + 63) / 64, 1);
    lin_wmma_kernel<<<gl, 256, 0, stream>>>(h2, linf, lb + i * ODIM_, xcat, i);
  }
  for (int e = 0; e < NEXP_; ++e)
    pack_b_rowmajor<<<(16 * KTEXP_ * 512 + 255) / 256, 256, 0, stream>>>(
        ew + (size_t)e * ODIM_ * KEXP_, expfr + (size_t)e * 16 * KTEXP_ * 512,
        KEXP_, KTEXP_);
  gate_kernel<<<(NTOK_ * NEXP_ + 255) / 256, 256, 0, stream>>>(xcat, gw, gb,
                                                               logits);
  topk_kernel<<<(NTOK_ + 255) / 256, 256, 0, stream>>>(logits, Wsel);
  dim3 ge((NTOK_ + 31) / 32, 1);
  expert_wmma_kernel<<<ge, 256, 0, stream>>>(xcat, expfr, ebp, Wsel, out);
  posemb_kernel<<<((2 * T2_ - 1) * ODIM_ + 255) / 256, 256, 0, stream>>>(pe);
  mask_kernel<<<(B_ * T2_ + 255) / 256, 256, 0, stream>>>(xmask, mo);
}
